// CharRNN_78391743087073
// MI455X (gfx1250) — matrix operand order repaired and checked by reading, not compiled, not run on hardware
//
#include <hip/hip_runtime.h>
#include <hip/hip_bf16.h>
#include <math.h>

typedef __bf16 bf16_t;
typedef __attribute__((ext_vector_type(16))) __bf16 bf16x16;
typedef __attribute__((ext_vector_type(8)))  float  f32x8;

union FragU { uint4 u[2]; bf16x16 v; };

// Fixed problem dimensions (from reference setup_inputs)
constexpr int CB  = 32;
constexpr int CT  = 256;
constexpr int CV  = 32000;
constexpr int CD  = 512;
constexpr int CBT = CB * CT;        // 8192 rows
constexpr int GK  = 2 * CD;         // 1024  gates GEMM K
constexpr int GN  = 3 * CD;         // 1536  gates GEMM N
#define NSPLIT 4

// ---------------------------------------------------------------------------
// Fragment loaders (CDNA5 wave32 WMMA 16x16x32 bf16 layouts, ISA 7.12.2)
// A (16xK row-major): lane m(0-15): K k0..k0+7 (v0-3) and k0+16..23 (v4-7);
//                     lane m+16:    K k0+8..15        and k0+24..31
// B (as Bt, NxK row-major): same K split as A (lane n: k0..7,k0+16..23; lane n+16: k0+8..15,k0+24..31)
// ---------------------------------------------------------------------------
__device__ __forceinline__ bf16x16 load_frag_a_row(const bf16_t* __restrict__ base,
                                                   int ld, int row0, int k0, int lane) {
  int half = lane >> 4;
  int lm = lane & 15;
  const bf16_t* p = base + (size_t)(row0 + lm) * ld + k0 + half * 8;
  FragU f;
  f.u[0] = *(const uint4*)(p);
  f.u[1] = *(const uint4*)(p + 16);
  return f.v;
}

__device__ __forceinline__ bf16x16 load_frag_b_row(const bf16_t* __restrict__ base,
                                                   int ld, int nrow, int k0, int lane) {
  int half = lane >> 4;
  const bf16_t* p = base + (size_t)nrow * ld + k0 + half * 8;
  FragU f;
  f.u[0] = *(const uint4*)(p);
  f.u[1] = *(const uint4*)(p + 16);
  return f.v;
}

// ---------------------------------------------------------------------------
// softmax_w (D x V, f32) -> WtV (V x D, bf16) via LDS-tiled transpose
// ---------------------------------------------------------------------------
__global__ void k_transpose_sw(const float* __restrict__ sw, bf16_t* __restrict__ Wt) {
  __shared__ float tile[32][33];
  int v0 = blockIdx.x * 32, d0 = blockIdx.y * 32;
  int tx = threadIdx.x, ty = threadIdx.y;            // (32, 8)
  for (int yy = ty; yy < 32; yy += 8)
    tile[yy][tx] = sw[(size_t)(d0 + yy) * CV + v0 + tx];
  __syncthreads();
  for (int yy = ty; yy < 32; yy += 8)
    Wt[(size_t)(v0 + yy) * CD + d0 + tx] = (bf16_t)tile[tx][yy];
}

// ---------------------------------------------------------------------------
// Embedding gather -> bf16
// ---------------------------------------------------------------------------
__global__ void k_embed(const int* __restrict__ ids, const float* __restrict__ emb,
                        bf16_t* __restrict__ X) {
  int i = blockIdx.x * blockDim.x + threadIdx.x;
  if (i >= CBT * CD) return;
  int r = i / CD, d = i % CD;
  X[i] = (bf16_t)emb[(size_t)ids[r] * CD + d];
}

// ---------------------------------------------------------------------------
// Pack gate weights: Wt[(g*D+e)*2D + k] = W{g}[k<D?0:1][k%D][e], and bias
// ---------------------------------------------------------------------------
__global__ void k_pack_gates(const float* __restrict__ Wz, const float* __restrict__ Wf,
                             const float* __restrict__ Wo, const float* __restrict__ bz,
                             const float* __restrict__ bfv, const float* __restrict__ bo,
                             bf16_t* __restrict__ Wt, float* __restrict__ biasp) {
  int i = blockIdx.x * blockDim.x + threadIdx.x;
  if (i >= GN * GK) return;
  int k = i % GK;
  int eg = i / GK;
  int g = eg / CD, e = eg % CD;
  const float* W = (g == 0) ? Wz : (g == 1) ? Wf : Wo;
  int s = k / CD, kk = k % CD;
  Wt[i] = (bf16_t)W[((size_t)s * CD + kk) * CD + e];
  if (k == 0) biasp[eg] = ((g == 0) ? bz : (g == 1) ? bfv : bo)[e];
}

// ---------------------------------------------------------------------------
// Apack[r] = [ x_prev[r] | x[r] ]  (x_prev = x shifted by one step in T)
// ---------------------------------------------------------------------------
__global__ void k_shiftcat(const bf16_t* __restrict__ X, bf16_t* __restrict__ Apack) {
  int i = blockIdx.x * blockDim.x + threadIdx.x;
  if (i >= CBT * CD) return;
  int r = i / CD, d = i % CD;
  int t = r % CT;
  Apack[(size_t)r * GK + CD + d] = X[i];
  Apack[(size_t)r * GK + d] = (t > 0) ? X[i - CD] : (bf16_t)0.0f;
}

// ---------------------------------------------------------------------------
// pre(M x GN f32) = Apack(M x GK bf16) * WtG(GN x GK bf16)^T + bias[n]
// block: 256 thr = 8 waves; wave tile 16x64; block tile 64x128
// Ping-pong double-buffered (k step 64): no fragment rotation moves.
// ---------------------------------------------------------------------------
__global__ __launch_bounds__(256) void k_gemm_bias(
    const bf16_t* __restrict__ A, const bf16_t* __restrict__ Bt,
    const float* __restrict__ bias, float* __restrict__ C) {
  int wave = threadIdx.x >> 5;
  int lane = threadIdx.x & 31;
  int mi = wave & 3;
  int ng = wave >> 2;
  int m0 = blockIdx.x * 64 + mi * 16;
  int n0 = blockIdx.y * 128 + ng * 64;
  int lm = lane & 15;

  f32x8 acc[4] = {};
  bf16x16 a0 = load_frag_a_row(A, GK, m0, 0, lane);
  bf16x16 b0[4];
#pragma unroll
  for (int t = 0; t < 4; ++t)
    b0[t] = load_frag_b_row(Bt, GK, n0 + 16 * t + lm, 0, lane);

  for (int k0 = 0; k0 < GK; k0 += 64) {
    int k1 = k0 + 32;
    bf16x16 a1 = load_frag_a_row(A, GK, m0, k1, lane);
    bf16x16 b1[4];
#pragma unroll
    for (int t = 0; t < 4; ++t)
      b1[t] = load_frag_b_row(Bt, GK, n0 + 16 * t + lm, k1, lane);
#pragma unroll
    for (int t = 0; t < 4; ++t)
      acc[t] = __builtin_amdgcn_wmma_f32_16x16x32_bf16(false, a0, false, b0[t],
                                                       (short)0, acc[t], false, false);
    int k2 = (k0 + 64 < GK) ? (k0 + 64) : 0;   // last iter: harmless reload of k=0
    a0 = load_frag_a_row(A, GK, m0, k2, lane);
#pragma unroll
    for (int t = 0; t < 4; ++t)
      b0[t] = load_frag_b_row(Bt, GK, n0 + 16 * t + lm, k2, lane);
#pragma unroll
    for (int t = 0; t < 4; ++t)
      acc[t] = __builtin_amdgcn_wmma_f32_16x16x32_bf16(false, a1, false, b1[t],
                                                       (short)0, acc[t], false, false);
  }

  int row = (lane < 16) ? 0 : 8;
#pragma unroll
  for (int t = 0; t < 4; ++t) {
    int n = n0 + 16 * t + lm;
    float bv = bias[n];
#pragma unroll
    for (int j = 0; j < 8; ++j)
      C[(size_t)(m0 + row + j) * GN + n] = acc[t][j] + bv;
  }
}

// ---------------------------------------------------------------------------
// Fused activations + fo-pool scan:  c = f*c + (1-f)*z ; h = o*c  -> bf16
// ---------------------------------------------------------------------------
__global__ void k_scan(const float* __restrict__ pre, bf16_t* __restrict__ H) {
  int i = blockIdx.x * blockDim.x + threadIdx.x;
  if (i >= CB * CD) return;
  int b = i / CD, d = i % CD;
  float c = 0.0f;
  for (int t = 0; t < CT; ++t) {
    size_t rbase = (size_t)(b * CT + t) * GN;
    float pz = pre[rbase + d];
    float pf = pre[rbase + CD + d];
    float po = pre[rbase + 2 * CD + d];
    float z = tanhf(pz);
    float f = 1.0f / (1.0f + __expf(-pf));
    float o = 1.0f / (1.0f + __expf(-po));
    c = f * c + (1.0f - f) * z;
    H[(size_t)(b * CT + t) * CD + d] = (bf16_t)(o * c);
  }
}

// ---------------------------------------------------------------------------
// Fused logits + online log-sum-exp.
// Block: 32 rows (two 16-row A tiles in LDS -> each B fragment feeds 2 WMMAs,
// halving B-side L2 traffic), 8 waves x 64 cols per 512-col chunk,
// grid.y = V split. Next-chunk B rows prefetched during the K sweep.
// partial[sp*BT + r] = (runMax, runSumExp) over this split's columns.
// ---------------------------------------------------------------------------
__global__ __launch_bounds__(256) void k_lse(
    const bf16_t* __restrict__ H, const bf16_t* __restrict__ WtV,
    const float* __restrict__ sb, float* __restrict__ partial) {
  __shared__ bf16_t As[32 * 520];          // padded row stride: conflict-light b128
  __shared__ float stats[8][32][2];
  int lane = threadIdx.x & 31, wave = threadIdx.x >> 5;
  int lm = lane & 15;
  int half = lane >> 4;
  int m0 = blockIdx.x * 32;
  int sp = blockIdx.y;
  int vStart = sp * (CV / NSPLIT);
  int vEnd = vStart + CV / NSPLIT;

  // stage 32xD bf16 A tile into LDS (16B chunks)
  for (int idx = threadIdx.x; idx < 32 * (CD / 8); idx += 256) {
    int r = idx >> 6;          // D/8 = 64 chunks per row
    int cc = idx & 63;
    *(uint4*)(&As[r * 520 + cc * 8]) = *(const uint4*)(&H[(size_t)(m0 + r) * CD + cc * 8]);
  }
  __syncthreads();

  float rm[2][8], rs[2][8];
#pragma unroll
  for (int i = 0; i < 2; ++i)
#pragma unroll
    for (int j = 0; j < 8; ++j) { rm[i][j] = -3.0e38f; rs[i][j] = 0.0f; }

  for (int cb = vStart; cb < vEnd; cb += 512) {
    int n0w = cb + wave * 64;
    f32x8 acc[2][4] = {};
    for (int k0 = 0; k0 < CD; k0 += 32) {
      FragU fa[2];
#pragma unroll
      for (int i = 0; i < 2; ++i) {
        const bf16_t* ap = &As[(i * 16 + lm) * 520 + k0 + half * 8];
        fa[i].u[0] = *(const uint4*)(ap);
        fa[i].u[1] = *(const uint4*)(ap + 16);
      }
#pragma unroll
      for (int t = 0; t < 4; ++t) {
        int nrow = n0w + 16 * t + lm;
        nrow = (nrow < CV - 1) ? nrow : (CV - 1);     // clamp OOB tail loads
        bf16x16 b = load_frag_b_row(WtV, CD, nrow, k0, lane);
        // pull next chunk's B rows toward the WGP while this chunk computes
        int npf = nrow + 512;
        npf = (npf < CV - 1) ? npf : (CV - 1);
        __builtin_prefetch(WtV + (size_t)npf * CD + k0, 0, 1);
        acc[0][t] = __builtin_amdgcn_wmma_f32_16x16x32_bf16(false, fa[0].v, false, b,
                                                            (short)0, acc[0][t], false, false);
        acc[1][t] = __builtin_amdgcn_wmma_f32_16x16x32_bf16(false, fa[1].v, false, b,
                                                            (short)0, acc[1][t], false, false);
      }
    }
    // online (max, sumexp) per row; rows split lanes 0-15 / 16-31 per VGPR j
    float bv[4]; bool ok[4];
#pragma unroll
    for (int t = 0; t < 4; ++t) {
      int ncol = n0w + 16 * t + lm;
      ok[t] = (ncol < vEnd);
      bv[t] = sb[(ncol < CV - 1) ? ncol : (CV - 1)];
    }
#pragma unroll
    for (int i = 0; i < 2; ++i) {
#pragma unroll
      for (int j = 0; j < 8; ++j) {
        float v[4];
#pragma unroll
        for (int t = 0; t < 4; ++t)
          v[t] = ok[t] ? (acc[i][t][j] + bv[t]) : -3.0e38f;
        float mloc = fmaxf(fmaxf(v[0], v[1]), fmaxf(v[2], v[3]));
#pragma unroll
        for (int msk = 1; msk < 16; msk <<= 1)
          mloc = fmaxf(mloc, __shfl_xor(mloc, msk, 32));
        float sloc = 0.0f;
#pragma unroll
        for (int t = 0; t < 4; ++t)
          sloc += ok[t] ? __expf(v[t] - mloc) : 0.0f;
#pragma unroll
        for (int msk = 1; msk < 16; msk <<= 1)
          sloc += __shfl_xor(sloc, msk, 32);
        float nm = fmaxf(rm[i][j], mloc);
        rs[i][j] = rs[i][j] * __expf(rm[i][j] - nm) + sloc * __expf(mloc - nm);
        rm[i][j] = nm;
      }
    }
  }

  if (lane == 0 || lane == 16) {
    int base = half * 8;
#pragma unroll
    for (int i = 0; i < 2; ++i)
#pragma unroll
      for (int j = 0; j < 8; ++j) {
        stats[wave][i * 16 + base + j][0] = rm[i][j];
        stats[wave][i * 16 + base + j][1] = rs[i][j];
      }
  }
  __syncthreads();
  if (threadIdx.x < 32) {
    int row = threadIdx.x;
    float m = -3.0e38f, s = 0.0f;
    for (int w = 0; w < 8; ++w) {
      float pm = stats[w][row][0], ps = stats[w][row][1];
      float nm = fmaxf(m, pm);
      s = s * __expf(m - nm) + ps * __expf(pm - nm);
      m = nm;
    }
    size_t o = ((size_t)sp * CBT + m0 + row) * 2;
    partial[o] = m;
    partial[o + 1] = s;
  }
}

// ---------------------------------------------------------------------------
// Per-row NLL: merge split partials + target-logit dot (1 wave per row)
// ---------------------------------------------------------------------------
__global__ __launch_bounds__(256) void k_nll(
    const bf16_t* __restrict__ H, const bf16_t* __restrict__ WtV,
    const float* __restrict__ sb, const int* __restrict__ tgt,
    const float* __restrict__ partial, float* __restrict__ nll) {
  int wave = threadIdx.x >> 5, lane = threadIdx.x & 31;
  int r = blockIdx.x * 8 + wave;
  if (r >= CBT) return;
  int tg = tgt[r];
  float dot = 0.0f;
  for (int k = lane; k < CD; k += 32)
    dot += (float)H[(size_t)r * CD + k] * (float)WtV[(size_t)tg * CD + k];
#pragma unroll
  for (int msk = 16; msk; msk >>= 1) dot += __shfl_xor(dot, msk, 32);
  if (lane == 0) {
    float logit_t = dot + sb[tg];
    float m = -3.0e38f, s = 0.0f;
    for (int sp = 0; sp < NSPLIT; ++sp) {
      float pm = partial[((size_t)sp * CBT + r) * 2];
      float ps = partial[((size_t)sp * CBT + r) * 2 + 1];
      float nm = fmaxf(m, pm);
      s = s * __expf(m - nm) + ps * __expf(pm - nm);
      m = nm;
    }
    nll[r] = (m + __logf(s)) - logit_t;
  }
}

__global__ void k_mean(const float* __restrict__ nll, float* __restrict__ out) {
  __shared__ float red[256];
  float s = 0.0f;
  for (int i = threadIdx.x; i < CBT; i += 256) s += nll[i];
  red[threadIdx.x] = s;
  __syncthreads();
  for (int offv = 128; offv; offv >>= 1) {
    if ((int)threadIdx.x < offv) red[threadIdx.x] += red[threadIdx.x + offv];
    __syncthreads();
  }
  if (threadIdx.x == 0) out[0] = red[0] / (float)CBT;
}

// ---------------------------------------------------------------------------
extern "C" void kernel_launch(void* const* d_in, const int* in_sizes, int n_in,
                              void* d_out, int out_size, void* d_ws, size_t ws_size,
                              hipStream_t stream) {
  (void)in_sizes; (void)n_in; (void)out_size; (void)ws_size;
  const int* ids = (const int*)d_in[0];
  const int* tgt = (const int*)d_in[1];
  const float* emb = (const float*)d_in[2];
  const float* Wz[2] = {(const float*)d_in[3],  (const float*)d_in[9]};
  const float* bz[2] = {(const float*)d_in[4],  (const float*)d_in[10]};
  const float* Wf[2] = {(const float*)d_in[5],  (const float*)d_in[11]};
  const float* bfv[2]= {(const float*)d_in[6],  (const float*)d_in[12]};
  const float* Wo[2] = {(const float*)d_in[7],  (const float*)d_in[13]};
  const float* bo[2] = {(const float*)d_in[8],  (const float*)d_in[14]};
  const float* sw = (const float*)d_in[15];
  const float* sb = (const float*)d_in[16];

  char* ws = (char*)d_ws;
  size_t off = 0;
  auto alloc = [&](size_t bytes) -> void* {
    void* p = ws + off;
    off = (off + bytes + 255) & ~(size_t)255;
    return p;
  };
  bf16_t* X     = (bf16_t*)alloc((size_t)CBT * CD * 2);         // layer I/O (reused)
  bf16_t* Apack = (bf16_t*)alloc((size_t)CBT * GK * 2);         // [x_prev|x]
  bf16_t* WtG   = (bf16_t*)alloc((size_t)GN * GK * 2);          // packed gate weights
  float*  biasG = (float*) alloc((size_t)GN * 4);
  float*  pre   = (float*) alloc((size_t)CBT * GN * 4);         // gate pre-activations
  bf16_t* WtV   = (bf16_t*)alloc((size_t)CV * CD * 2);          // softmax_w^T bf16
  float*  part  = (float*) alloc((size_t)NSPLIT * CBT * 2 * 4); // (max,sumexp) partials
  float*  nllb  = (float*) alloc((size_t)CBT * 4);

  k_transpose_sw<<<dim3(CV / 32, CD / 32), dim3(32, 8), 0, stream>>>(sw, WtV);
  k_embed<<<(CBT * CD + 255) / 256, 256, 0, stream>>>(ids, emb, X);

  for (int l = 0; l < 2; ++l) {
    k_pack_gates<<<(GN * GK + 255) / 256, 256, 0, stream>>>(
        Wz[l], Wf[l], Wo[l], bz[l], bfv[l], bo[l], WtG, biasG);
    k_shiftcat<<<(CBT * CD + 255) / 256, 256, 0, stream>>>(X, Apack);
    k_gemm_bias<<<dim3(CBT / 64, GN / 128), 256, 0, stream>>>(Apack, WtG, biasG, pre);
    k_scan<<<(CB * CD + 255) / 256, 256, 0, stream>>>(pre, X);
  }

  k_lse<<<dim3(CBT / 32, NSPLIT), 256, 0, stream>>>(X, WtV, sb, part);
  k_nll<<<(CBT + 7) / 8, 256, 0, stream>>>(X, WtV, sb, tgt, part, nllb);
  k_mean<<<1, 256, 0, stream>>>(nllb, (float*)d_out);
}
